// SpikingNeuralNetwork_16776142259105
// MI455X (gfx1250) — matrix operand order repaired and checked by reading, not compiled, not run on hardware
//
#include <hip/hip_runtime.h>
#include <hip/hip_bf16.h>
#include <cstdint>

// ---------------------------------------------------------------------------
// Spiking NN scan for MI455X (gfx1250, wave32, WMMA).
//   B=64, T=128, I=512, H=2048, O=512
// Phase 1: cur[B*T,H] = bf16(x) @ bf16(W_in)      (fully parallel, wmma bf16)
// Phase 2: persistent scan, 4 row-groups x 8 column-slice WGs,
//          spk@L and spk@W_h as wmma bf16, atomic inter-WG barrier,
//          spike block staged into LDS via GLOBAL_LOAD_ASYNC_TO_LDS_B128.
// ---------------------------------------------------------------------------

typedef __attribute__((ext_vector_type(16))) __bf16 v16bf;
typedef __attribute__((ext_vector_type(8)))  __bf16 v8bf;
typedef __attribute__((ext_vector_type(8)))  float  v8f;
typedef __attribute__((ext_vector_type(4)))  int    v4i;

#define BDIM  64
#define TDIM  128
#define IDIM  512
#define HDIM  2048
#define ODIM  512

#define AS3 __attribute__((address_space(3)))

#if defined(__gfx1250__) && \
    __has_builtin(__builtin_amdgcn_global_load_async_to_lds_b128) && \
    __has_builtin(__builtin_amdgcn_s_wait_asynccnt)
#define USE_ASYNC_LDS 1
#else
#define USE_ASYNC_LDS 0
#endif

static __device__ __forceinline__ v16bf cat8(v8bf lo, v8bf hi) {
  return __builtin_shufflevector(lo, hi, 0,1,2,3,4,5,6,7,8,9,10,11,12,13,14,15);
}

static __device__ __forceinline__ v8f wmma_bf16(v16bf a, v16bf b, v8f c) {
  // D = A(16x32 bf16) * B(32x16 bf16) + C(16x16 f32)
  return __builtin_amdgcn_wmma_f32_16x16x32_bf16(false, a, false, b,
                                                 (short)0, c, false, false);
}

// A-fragment (16x32 bf16): lane<16 -> K {0..15}, lane>=16 -> K {16..31}.
static __device__ __forceinline__ v16bf loadA(const __bf16* row, int k0, int lh) {
  v8bf lo = *(const v8bf*)(row + k0 + 16 * lh);
  v8bf hi = *(const v8bf*)(row + k0 + 16 * lh + 8);
  return cat8(lo, hi);
}

// --------------------------- conversion kernels ----------------------------

__global__ void cvt_plain_kernel(const float* __restrict__ src,
                                 __bf16* __restrict__ dst, int n) {
  int i = blockIdx.x * blockDim.x + threadIdx.x;
  if (i < n) dst[i] = (__bf16)src[i];
}

// dst[n*K + k] = bf16(scale * src[k*N + n])   (transpose K x N -> N x K)
__global__ void cvt_transpose_kernel(const float* __restrict__ src,
                                     __bf16* __restrict__ dst,
                                     int K, int N, float scale) {
  long long i = (long long)blockIdx.x * blockDim.x + threadIdx.x;
  if (i < (long long)K * N) {
    int n = (int)(i / K);
    int k = (int)(i % K);
    dst[i] = (__bf16)(scale * src[(long long)k * N + n]);
  }
}

// --------------------- Phase 1: cur = Xb @ W_in (bf16) ---------------------
// Software-pipelined: fragments for k+32 are fetched before the k WMMAs issue.
__global__ void gemm_cur_kernel(const __bf16* __restrict__ Xb,
                                const __bf16* __restrict__ WtT,
                                float* __restrict__ cur) {
  const int tid  = threadIdx.x;
  const int wave = tid >> 5;
  const int lane = tid & 31;
  const int lm   = lane & 15;
  const int lh   = lane >> 4;
  const int wm   = wave >> 2;      // 0..1
  const int wn   = wave & 3;       // 0..3
  const int m0   = blockIdx.x * 32 + wm * 16;
  const int n0   = blockIdx.y * 256 + wn * 64;

  const __bf16* arow = Xb + (long long)(m0 + lm) * IDIM;
  const __bf16* brow[4];
#pragma unroll
  for (int j = 0; j < 4; ++j)
    brow[j] = WtT + (long long)(n0 + 16 * j + lm) * IDIM + 16 * lh;

  v8f acc[4] = {};
  v16bf a = loadA(arow, 0, lh);
  v16bf b[4];
#pragma unroll
  for (int j = 0; j < 4; ++j) b[j] = *(const v16bf*)(brow[j]);

  for (int k0 = 0; k0 < IDIM; k0 += 32) {
    const int kn = (k0 + 32) & (IDIM - 1);        // wrap on last iter
    v16bf an = loadA(arow, kn, lh);
    v16bf bn[4];
#pragma unroll
    for (int j = 0; j < 4; ++j) {
      bn[j] = *(const v16bf*)(brow[j] + kn);
      __builtin_prefetch(brow[j] + kn + 64, 0, 3);
    }
#pragma unroll
    for (int j = 0; j < 4; ++j) acc[j] = wmma_bf16(a, b[j], acc[j]);
    a = an;
#pragma unroll
    for (int j = 0; j < 4; ++j) b[j] = bn[j];
  }

#pragma unroll
  for (int j = 0; j < 4; ++j) {
#pragma unroll
    for (int r = 0; r < 8; ++r) {
      int m = m0 + r + 8 * lh;       // C/D layout: lanes>=16 hold M=8..15
      int n = n0 + 16 * j + lm;
      cur[(long long)m * HDIM + n] = acc[j][r];
    }
  }
}

// --------------------------- Phase 2: the scan -----------------------------
__global__ void snn_scan_kernel(const float* __restrict__ cur,
                                const __bf16* __restrict__ LbT,   // [H][H] = -0.1*L^T
                                const __bf16* __restrict__ HbT,   // [O][H] = W_h^T
                                float* __restrict__ out,
                                __bf16* __restrict__ spk,         // [2][4][16][H]
                                unsigned* __restrict__ ctr) {     // [4]
  const int g = blockIdx.x >> 3;   // row group 0..3
  const int p = blockIdx.x & 7;    // column part 0..7
  const int tid  = threadIdx.x;
  const int wave = tid >> 5;
  const int lane = tid & 31;
  const int lm   = lane & 15;
  const int lh   = lane >> 4;

  const int row_ = tid & 15;            // state row this thread owns
  const int cb   = (tid >> 4) * 16;     // state col base within 256-slice

  float mp[16];
  float rc[16];
#pragma unroll
  for (int i = 0; i < 16; ++i) { mp[i] = 0.f; rc[i] = 0.f; }

  __shared__ __bf16 spkA[16 * HDIM];    // staged spike block  (64 KB)
  __shared__ float  slab[16][256];      // lateral-product handoff (16 KB)

  for (int t = 0; t < TDIM; ++t) {
    const int par = t & 1;
    const long long spkBase = (long long)((par * 4 + g) * 16) * HDIM;

    // ---- 1. integrate, threshold, refractory; emit spike slice -----------
    const float* curp = cur + (((long long)(g * 16 + row_) * TDIM) + t) * HDIM
                            + p * 256 + cb;
    v16bf sv;
#pragma unroll
    for (int i = 0; i < 16; ++i) {
      float c      = curp[i];
      float active = (rc[i] <= 0.f) ? 1.f : 0.f;
      mp[i]        = 0.95f * mp[i] + c * active;
      float s      = (mp[i] >= 1.f) ? 1.f : 0.f;
      mp[i]       *= (1.f - s);
      rc[i]        = fmaxf(rc[i] - 1.f, 0.f) + s * 2.f;
      sv[i]        = (__bf16)s;
    }
    *(v16bf*)(spk + spkBase + (long long)row_ * HDIM + p * 256 + cb) = sv;

    // ---- 2. release stores, barrier across the 8 WGs of this group -------
    __threadfence();
    __syncthreads();
    if (tid == 0) {
      __hip_atomic_fetch_add(&ctr[g], 1u, __ATOMIC_RELEASE,
                             __HIP_MEMORY_SCOPE_AGENT);
      const unsigned tgt = 8u * (unsigned)(t + 1);
      while (__hip_atomic_load(&ctr[g], __ATOMIC_ACQUIRE,
                               __HIP_MEMORY_SCOPE_AGENT) < tgt) {
        __builtin_amdgcn_s_sleep(2);
      }
    }
    __syncthreads();
    __threadfence();   // acquire: don't read stale spike lines

    // ---- 3. stage full 16x2048 spike block into LDS ----------------------
    {
      const __bf16* srcg = spk + spkBase;
#if USE_ASYNC_LDS
#pragma unroll
      for (int i = 0; i < 16; ++i) {
        const unsigned e = (unsigned)tid * 8u + (unsigned)i * 2048u;
        __builtin_amdgcn_global_load_async_to_lds_b128(
            (v4i*)(srcg + e),
            (AS3 v4i*)(spkA + e), 0, 0);
      }
      __builtin_amdgcn_s_wait_asynccnt(0);
#else
      for (int i = tid; i < (16 * HDIM) / 8; i += 256)
        ((v8bf*)spkA)[i] = ((const v8bf*)srcg)[i];
#endif
      __syncthreads();
    }

    // ---- 4. lateral GEMM: [16,2048] @ LbT-slice -> [16,256] --------------
    const __bf16* arow = spkA + (long long)lm * HDIM;       // from LDS
    const int nb0 = p * 256 + wave * 32;
    const __bf16* bptr0 = LbT + (long long)(nb0 + lm) * HDIM + 16 * lh;
    const __bf16* bptr1 = LbT + (long long)(nb0 + 16 + lm) * HDIM + 16 * lh;

    v8f accL[2] = {};
    v16bf a  = loadA(arow, 0, lh);
    v16bf b0 = *(const v16bf*)(bptr0);
    v16bf b1 = *(const v16bf*)(bptr1);
    for (int k0 = 0; k0 < HDIM; k0 += 32) {
      const int kn = (k0 + 32) & (HDIM - 1);
      v16bf an  = loadA(arow, kn, lh);
      v16bf b0n = *(const v16bf*)(bptr0 + kn);
      v16bf b1n = *(const v16bf*)(bptr1 + kn);
      __builtin_prefetch(bptr0 + kn + 64, 0, 3);
      __builtin_prefetch(bptr1 + kn + 64, 0, 3);
      accL[0] = wmma_bf16(a, b0, accL[0]);
      accL[1] = wmma_bf16(a, b1, accL[1]);
      a = an; b0 = b0n; b1 = b1n;
    }
#pragma unroll
    for (int j = 0; j < 2; ++j) {
#pragma unroll
      for (int r = 0; r < 8; ++r) {
        slab[r + 8 * lh][wave * 32 + 16 * j + lm] = accL[j][r];
      }
    }

    // ---- 5. output GEMM (waves 0..3): [16,2048] @ HbT-slice -> [16,64] ---
    if (wave < 4) {
      const int n = p * 64 + wave * 16 + lm;
      const __bf16* hptr = HbT + (long long)n * HDIM + 16 * lh;
      v8f accH = {};
      v16bf ah = loadA(arow, 0, lh);
      v16bf bh = *(const v16bf*)(hptr);
      for (int k0 = 0; k0 < HDIM; k0 += 32) {
        const int kn = (k0 + 32) & (HDIM - 1);
        v16bf ahn = loadA(arow, kn, lh);
        v16bf bhn = *(const v16bf*)(hptr + kn);
        accH = wmma_bf16(ah, bh, accH);
        ah = ahn; bh = bhn;
      }
#pragma unroll
      for (int r = 0; r < 8; ++r) {
        int m = r + 8 * lh;
        long long oidx = (((long long)(g * 16 + m) * TDIM) + t) * ODIM + n;
        out[oidx] = accH[r];
      }
    }
    __syncthreads();

    // ---- 6. apply (pre-negated, pre-scaled) lateral product to mp --------
#pragma unroll
    for (int i = 0; i < 16; ++i) mp[i] += slab[row_][cb + i];
    // slab/spkA are only rewritten after the next step's inter-WG barrier
    // (which contains __syncthreads), so no extra barrier needed here.
  }
}

// ------------------------------- launcher ----------------------------------

extern "C" void kernel_launch(void* const* d_in, const int* in_sizes, int n_in,
                              void* d_out, int out_size, void* d_ws, size_t ws_size,
                              hipStream_t stream) {
  const float* x    = (const float*)d_in[0];   // [B,T,I]
  const float* w_in = (const float*)d_in[1];   // [I,H]
  const float* w_h  = (const float*)d_in[2];   // [H,O]
  const float* lat  = (const float*)d_in[3];   // [H,H]
  float* out = (float*)d_out;                  // [B,T,O]

  char* ws = (char*)d_ws;
  // workspace layout (256B-aligned regions)
  unsigned* ctr = (unsigned*)(ws + 0);                          // counters
  float*  cur = (float*) (ws + 256);                            // 64 MB
  __bf16* Xb  = (__bf16*)(ws + 256 + 67108864LL);               // 8 MB
  __bf16* WtT = (__bf16*)(ws + 256 + 67108864LL + 8388608LL);   // 2 MB
  __bf16* LbT = (__bf16*)(ws + 256 + 67108864LL + 8388608LL
                              + 2097152LL);                     // 8 MB
  __bf16* HbT = (__bf16*)(ws + 256 + 67108864LL + 8388608LL
                              + 2097152LL + 8388608LL);         // 2 MB
  __bf16* spk = (__bf16*)(ws + 256 + 67108864LL + 8388608LL
                              + 2097152LL + 8388608LL + 2097152LL); // 512 KB

  // zero the barrier counters every launch (graph-capture safe)
  (void)hipMemsetAsync(ws, 0, 256, stream);

  // bf16 conversions / transposes
  {
    int n = BDIM * TDIM * IDIM;
    cvt_plain_kernel<<<(n + 255) / 256, 256, 0, stream>>>(x, Xb, n);
  }
  {
    int n = IDIM * HDIM;   // W_in^T : [H][I]
    cvt_transpose_kernel<<<(n + 255) / 256, 256, 0, stream>>>(w_in, WtT,
                                                              IDIM, HDIM, 1.0f);
  }
  {
    int n = HDIM * HDIM;   // -0.1 * L^T : [H][H]
    cvt_transpose_kernel<<<(n + 255) / 256, 256, 0, stream>>>(lat, LbT,
                                                              HDIM, HDIM, -0.1f);
  }
  {
    int n = HDIM * ODIM;   // W_h^T : [O][H]
    cvt_transpose_kernel<<<(n + 255) / 256, 256, 0, stream>>>(w_h, HbT,
                                                              HDIM, ODIM, 1.0f);
  }

  // Phase 1: all input currents in one parallel GEMM
  {
    dim3 grid((BDIM * TDIM) / 32, HDIM / 256);
    gemm_cur_kernel<<<grid, 256, 0, stream>>>(Xb, WtT, cur);
  }

  // Phase 2: persistent sequential scan (32 cooperating WGs)
  snn_scan_kernel<<<32, 256, 0, stream>>>(cur, LbT, HbT, out, spk, ctr);
}
